// MultiSimilarityLoss_31671088840965
// MI455X (gfx1250) — matrix operand order repaired and checked by reading, not compiled, not run on hardware
//
#include <hip/hip_runtime.h>
#include <hip/hip_fp16.h>
#include <hip/hip_bf16.h>

typedef __attribute__((ext_vector_type(16))) _Float16 v16h;
typedef __attribute__((ext_vector_type(8)))  float    v8f;

#define BTOT   8192
#define DDIM   128
#define NCLS   512
#define ALPHA_ 2.0f
#define BETA_  50.0f
#define LAMBDA_ 0.5f

// ---------------------------------------------------------------------------
// Zero the per-row accumulators, histogram, and final scalar accumulators.
// ---------------------------------------------------------------------------
__global__ void msl_zero(float* __restrict__ pos, float* __restrict__ neg,
                         int* __restrict__ hist, float* __restrict__ acc) {
  int i = blockIdx.x * blockDim.x + threadIdx.x;
  if (i < BTOT) { pos[i] = 0.0f; neg[i] = 0.0f; }
  if (i < NCLS) hist[i] = 0;
  if (i < 2)    acc[i]  = 0.0f;
}

// ---------------------------------------------------------------------------
// Row L2-normalize fp32 -> fp16.  One wave (32 lanes) per row, float4/lane.
// ---------------------------------------------------------------------------
__global__ __launch_bounds__(256)
void msl_normalize(const float* __restrict__ x, _Float16* __restrict__ eh) {
  const int lane = threadIdx.x & 31;
  const int row  = blockIdx.x * 8 + (threadIdx.x >> 5);
  const float4 v = *(const float4*)(x + (size_t)row * DDIM + lane * 4);
  float ss = v.x * v.x + v.y * v.y + v.z * v.z + v.w * v.w;
#pragma unroll
  for (int m = 16; m >= 1; m >>= 1) ss += __shfl_xor(ss, m, 32);
  const float inv = 1.0f / fmaxf(sqrtf(ss), 1e-12f);
  _Float16* o = eh + (size_t)row * DDIM + lane * 4;
  o[0] = (_Float16)(v.x * inv);
  o[1] = (_Float16)(v.y * inv);
  o[2] = (_Float16)(v.z * inv);
  o[3] = (_Float16)(v.w * inv);
}

// ---------------------------------------------------------------------------
// Class histogram (for exact pos/neg-existence validity per row).
// ---------------------------------------------------------------------------
__global__ void msl_hist(const int* __restrict__ labels, int* __restrict__ hist) {
  int i = blockIdx.x * blockDim.x + threadIdx.x;
  if (i < BTOT) atomicAdd(&hist[labels[i]], 1);
}

// ---------------------------------------------------------------------------
// B fragment loader: lane L -> column n = colbase+(L&15); for K-block kb,
// element e = K kb*32 + h*16 + e  (h = L>>4).  Row base is 32B-aligned, so
// each v16h is two global_load_b128.
// ---------------------------------------------------------------------------
__device__ __forceinline__ void msl_load_bfrag(const _Float16* __restrict__ eh,
                                               int n, int h, v16h b[4]) {
  const _Float16* brow = eh + (size_t)n * DDIM;
#pragma unroll
  for (int kb = 0; kb < 4; ++kb)
    b[kb] = *(const v16h*)(brow + kb * 32 + h * 16);
}

// ---------------------------------------------------------------------------
// Fused GEMM (sim = e * e^T via v_wmma_f32_16x16x32_f16) + masked-exp
// epilogue.  One wave owns a 16-row tile across a 512-column chunk.
// B fragments are double-buffered (2x unrolled column loop) so the next
// tile's 8 global_load_b128 overlap the current tile's WMMAs + exps.
//   A frag (16x32 f16): lane L -> row rowbase+(L&15); elems[0..15]=K kb*32+h*16+e
//                                                             (h = L>>4)
//   C      (16x16 f32): VGPR r, lane L -> M = r + 8*h, N = L&15
// ---------------------------------------------------------------------------
__global__ __launch_bounds__(256)
void msl_wmma_epilogue(const _Float16* __restrict__ eh,
                       const int* __restrict__ labels,
                       float* __restrict__ pos_out,
                       float* __restrict__ neg_out,
                       int tiles_per_chunk) {
  const int lane    = threadIdx.x & 31;
  const int wid     = threadIdx.x >> 5;
  const int h       = lane >> 4;
  const int ln      = lane & 15;
  const int rowbase = (blockIdx.x * 8 + wid) * 16;

  // Resident A fragments: this wave's 16 rows, full K = 128.
  v16h afrag[4];
  {
    const _Float16* arow = eh + (size_t)(rowbase + ln) * DDIM;
#pragma unroll
    for (int kb = 0; kb < 4; ++kb) {
      const _Float16* p = arow + kb * 32 + h * 16;
#pragma unroll
      for (int t = 0; t < 8; ++t) {
        afrag[kb][t]     = p[t];
        afrag[kb][8 + t] = p[8 + t];
      }
    }
  }

  // Labels of the 8 C-rows this lane owns (M = rowbase + 8h + r).
  int lm[8];
#pragma unroll
  for (int r = 0; r < 8; ++r) lm[r] = labels[rowbase + h * 8 + r];

  float pacc[8], nacc[8];
#pragma unroll
  for (int r = 0; r < 8; ++r) { pacc[r] = 0.0f; nacc[r] = 0.0f; }

  // 4 WMMAs + one-exp-per-element epilogue on a staged B buffer.
  auto process = [&](const v16h* __restrict__ b, int n, int lc) {
    v8f c = {};
#pragma unroll
    for (int kb = 0; kb < 4; ++kb) {
      c = __builtin_amdgcn_wmma_f32_16x16x32_f16(
          /*neg_a=*/false, afrag[kb], /*neg_b=*/false, b[kb],
          /*c_mod=*/(short)0, c, /*reuse_a=*/false, /*reuse_b=*/false);
    }
#pragma unroll
    for (int r = 0; r < 8; ++r) {
      const int  m    = rowbase + h * 8 + r;
      const bool same = (lm[r] == lc);
      const float x   = c[r] - LAMBDA_;
      const float t   = same ? (-ALPHA_ * x) : (BETA_ * x);
      const float e   = __expf(t);
      pacc[r] += (same && (m != n)) ? e : 0.0f;
      nacc[r] += same ? 0.0f : e;
    }
  };

  const int tile0 = blockIdx.y * tiles_per_chunk;

  // Software-pipelined, double-buffered column-tile loop (tiles_per_chunk even).
  v16h b0[4], b1[4];
  int n0  = tile0 * 16 + ln;
  int lc0 = labels[n0];
  msl_load_bfrag(eh, n0, h, b0);

  for (int jt = 0; jt < tiles_per_chunk; jt += 2) {
    // Stage tile jt+1 into b1 while b0 is consumed.
    const int n1  = (tile0 + jt + 1) * 16 + ln;
    const int lc1 = labels[n1];
    msl_load_bfrag(eh, n1, h, b1);
    __builtin_prefetch(eh + (size_t)(n1 + 32) * DDIM, 0, 1);  // tile jt+3
    process(b0, n0, lc0);

    // Stage tile jt+2 into b0 while b1 is consumed (clamped at the end).
    const int j2 = (jt + 2 < tiles_per_chunk) ? jt + 2 : jt;
    const int n2 = (tile0 + j2) * 16 + ln;
    const int lc2 = labels[n2];
    msl_load_bfrag(eh, n2, h, b0);
    process(b1, n1, lc1);

    n0 = n2; lc0 = lc2;
  }

  // Row sums live in 16 lanes of a half-wave: butterfly-reduce, then commit.
#pragma unroll
  for (int r = 0; r < 8; ++r) {
    float p = pacc[r], q = nacc[r];
#pragma unroll
    for (int m = 8; m >= 1; m >>= 1) {
      p += __shfl_xor(p, m, 16);
      q += __shfl_xor(q, m, 16);
    }
    if (ln == r) {
      const int row = rowbase + h * 8 + r;
      atomicAdd(&pos_out[row], p);
      atomicAdd(&neg_out[row], q);
    }
  }
}

// ---------------------------------------------------------------------------
// Per-row log1p + validity, reduced into acc[0]=sum, acc[1]=n_valid.
// ---------------------------------------------------------------------------
__global__ __launch_bounds__(256)
void msl_finalize(const float* __restrict__ pos, const float* __restrict__ neg,
                  const int* __restrict__ hist, const int* __restrict__ labels,
                  float* __restrict__ acc) {
  int i = blockIdx.x * blockDim.x + threadIdx.x;
  float v = 0.0f, cv = 0.0f;
  if (i < BTOT) {
    const int cnt = hist[labels[i]];
    const bool valid = (cnt >= 2) && (cnt < BTOT);
    if (valid) {
      v  = log1pf(pos[i]) * (1.0f / ALPHA_) + log1pf(neg[i]) * (1.0f / BETA_);
      cv = 1.0f;
    }
  }
#pragma unroll
  for (int m = 16; m >= 1; m >>= 1) {
    v  += __shfl_xor(v, m, 32);
    cv += __shfl_xor(cv, m, 32);
  }
  if ((threadIdx.x & 31) == 0) {
    atomicAdd(&acc[0], v);
    atomicAdd(&acc[1], cv);
  }
}

__global__ void msl_writeout(const float* __restrict__ acc, float* __restrict__ out) {
  out[0] = acc[0] / fmaxf(acc[1], 1.0f);
}

// ---------------------------------------------------------------------------
extern "C" void kernel_launch(void* const* d_in, const int* in_sizes, int n_in,
                              void* d_out, int out_size, void* d_ws, size_t ws_size,
                              hipStream_t stream) {
  const float* emb    = (const float*)d_in[0];  // [8192,128] fp32
  const int*   labels = (const int*)d_in[1];    // [8192] int32
  float*       out    = (float*)d_out;          // scalar

  char* ws = (char*)d_ws;
  _Float16* eh  = (_Float16*)ws;                                   // 2 MB
  float*    pos = (float*)(ws + (size_t)BTOT * DDIM * sizeof(_Float16));
  float*    neg = pos + BTOT;
  int*      hist = (int*)(neg + BTOT);
  float*    acc  = (float*)(hist + NCLS);

  msl_zero<<<(BTOT + 255) / 256, 256, 0, stream>>>(pos, neg, hist, acc);
  msl_normalize<<<BTOT / 8, 256, 0, stream>>>(emb, eh);
  msl_hist<<<(BTOT + 255) / 256, 256, 0, stream>>>(labels, hist);

  const int NCHUNK = 16;                             // column chunks of 512
  const int tiles_per_chunk = (BTOT / 16) / NCHUNK;  // 32 tiles of 16 cols
  dim3 grid(BTOT / 16 / 8, NCHUNK);                  // (64, 16), 8 waves/block
  msl_wmma_epilogue<<<grid, 256, 0, stream>>>(eh, labels, pos, neg, tiles_per_chunk);

  msl_finalize<<<(BTOT + 255) / 256, 256, 0, stream>>>(pos, neg, hist, labels, acc);
  msl_writeout<<<1, 1, 0, stream>>>(acc, out);
}
